// Attention_67577015435735
// MI455X (gfx1250) — matrix operand order repaired and checked by reading, not compiled, not run on hardware
//
#include <hip/hip_runtime.h>
#include <hip/hip_bf16.h>

// Problem constants
#define BB   16
#define NN   1024
#define EMB  128
#define HH   8
#define INNER 1024          // EMB*HH
#define QKVN 3072           // 3*INNER
#define SCALE 0.03125f      // INNER^-0.5
#define EPS 1e-5f

typedef __attribute__((ext_vector_type(16))) __bf16 v16bf;
typedef __attribute__((ext_vector_type(8)))  __bf16 v8bf;
typedef __attribute__((ext_vector_type(8)))  float  v8f;

#define WMMA_BF16(a, b, c) \
    __builtin_amdgcn_wmma_f32_16x16x32_bf16(false, (a), false, (b), (short)0, (c), false, false)

// Wait until at most `n` async (global<->LDS) ops remain outstanding.
#define ASYNC_WAIT(n) asm volatile("s_wait_asynccnt " #n ::: "memory")

static __device__ __forceinline__ v16bf cat8(v8bf lo, v8bf hi) {
    return __builtin_shufflevector(lo, hi, 0,1,2,3,4,5,6,7,8,9,10,11,12,13,14,15);
}

// A-fragment (16x32 bf16, M x K) for one lane: row pointer already offset to
// this lane's M-row and k-step. elems 0-15 -> k = 16*half+e.
static __device__ __forceinline__ v16bf load_afrag(const __bf16* rowk, int half) {
    v8bf lo = *(const v8bf*)(rowk + 16 * half);
    v8bf hi = *(const v8bf*)(rowk + 16 * half + 8);
    return cat8(lo, hi);
}

// Async DMA of one 16-byte chunk global -> LDS (ASYNCcnt-tracked, no VGPR data).
// LDS operand is the 32-bit LDS offset = low 32 bits of the generic address
// (ISA 10.2: LDS aperture addresses truncate to addr[31:0]).
static __device__ __forceinline__ void async_ld16(const __bf16* g, const __bf16* l) {
    unsigned       loff = (unsigned)(size_t)l;
    unsigned long long ga = (unsigned long long)(size_t)g;
    asm volatile("global_load_async_to_lds_b128 %0, %1, off"
                 :: "v"(loff), "v"(ga) : "memory");
}

// ---------------------------------------------------------------------------
// Kernel 1: LayerNorm (fp32 in) -> bf16 out. One wave per 128-wide row.
// ---------------------------------------------------------------------------
__global__ __launch_bounds__(32) void ln_kernel(const float* __restrict__ x,
                                                const float* __restrict__ gamma,
                                                const float* __restrict__ beta,
                                                __bf16* __restrict__ xb) {
    const int row  = blockIdx.x;
    const int lane = threadIdx.x;
    const float4 v = ((const float4*)(x + (size_t)row * EMB))[lane];
    float s  = v.x + v.y + v.z + v.w;
    float s2 = v.x*v.x + v.y*v.y + v.z*v.z + v.w*v.w;
    for (int off = 1; off < 32; off <<= 1) {
        s  += __shfl_xor(s,  off, 32);
        s2 += __shfl_xor(s2, off, 32);
    }
    const float mu   = s * (1.0f / EMB);
    const float var  = s2 * (1.0f / EMB) - mu * mu;
    const float rinv = rsqrtf(var + EPS);
    const int c = lane * 4;
    const float e[4] = {v.x, v.y, v.z, v.w};
    for (int j = 0; j < 4; ++j) {
        float y = (e[j] - mu) * rinv * gamma[c + j] + beta[c + j];
        xb[(size_t)row * EMB + c + j] = (__bf16)y;
    }
}

// ---------------------------------------------------------------------------
// Kernels 2a/2b: weight convert + transpose to bf16 so every WMMA B-fragment
// load is one contiguous 32-byte run per lane.
// ---------------------------------------------------------------------------
__global__ __launch_bounds__(256) void conv_wqkv(const float* __restrict__ w,
                                                 __bf16* __restrict__ wt) {
    int idx = blockIdx.x * 256 + threadIdx.x;          // wt[3072][128]
    if (idx < QKVN * EMB) {
        int n = idx >> 7, k = idx & 127;
        wt[idx] = (__bf16)w[(size_t)k * QKVN + n];
    }
}

__global__ __launch_bounds__(256) void conv_wproj(const float* __restrict__ w,
                                                  __bf16* __restrict__ wt) {
    int idx = blockIdx.x * 256 + threadIdx.x;          // wt[128][1024]
    if (idx < EMB * INNER) {
        int n = idx >> 10, k = idx & 1023;
        wt[idx] = (__bf16)w[(size_t)k * EMB + n];
    }
}

// ---------------------------------------------------------------------------
// Kernel 3: QKV GEMM [16384 x 128] @ [128 x 3072], 16x64 C-tile per wave:
// A-fragments loaded once, reused across 4 column tiles (16 WMMAs / wave).
// Scatter into per-(b,h) Q,K row-major and V transposed [d][n].
// ---------------------------------------------------------------------------
__global__ __launch_bounds__(32) void qkv_gemm(const __bf16* __restrict__ X,
                                               const __bf16* __restrict__ Wt,
                                               __bf16* __restrict__ q_ws,
                                               __bf16* __restrict__ k_ws,
                                               __bf16* __restrict__ vt_ws) {
    const int lane = threadIdx.x, half = lane >> 4, lm = lane & 15;
    const int nt = blockIdx.x;     // 0..47 (64-wide N tiles)
    const int mt = blockIdx.y;     // 0..1023
    const __bf16* arow = X + (size_t)(mt * 16 + lm) * EMB;
    v16bf afrag[4];
#pragma unroll
    for (int k0 = 0; k0 < 4; ++k0) afrag[k0] = load_afrag(arow + k0 * 32, half);

    v8f acc[4] = {{}, {}, {}, {}};
#pragma unroll
    for (int c = 0; c < 4; ++c) {
        const __bf16* brow = Wt + (size_t)(nt * 64 + c * 16 + lm) * EMB;
#pragma unroll
        for (int k0 = 0; k0 < 4; ++k0) {
            v16bf b = *(const v16bf*)(brow + k0 * 32 + 16 * half);
            acc[c] = WMMA_BF16(afrag[k0], b, acc[c]);
        }
    }
#pragma unroll
    for (int c = 0; c < 4; ++c) {
        for (int r = 0; r < 8; ++r) {
            int gm = mt * 16 + r + 8 * half;          // global M row
            int gn = nt * 64 + c * 16 + lm;           // global N col
            int bidx = gm >> 10, iseq = gm & 1023;
            int which = gn / INNER, rem = gn % INNER;
            int head = rem >> 7, d = rem & 127;
            __bf16 val = (__bf16)acc[c][r];
            size_t bh = (size_t)(bidx * HH + head);
            if (which == 0)      q_ws[(bh * NN + iseq) * EMB + d] = val;
            else if (which == 1) k_ws[(bh * NN + iseq) * EMB + d] = val;
            else                 vt_ws[(bh * EMB + d) * NN + iseq] = val;
        }
    }
}

// ---------------------------------------------------------------------------
// Kernel 4: flash attention. One wave per (b, h, 16-query tile).
// K panels (32 keys x 128 d = 8 KB) are double-buffered in LDS via
// GLOBAL_LOAD_ASYNC_TO_LDS_B128; the next panel's DMA overlaps the current
// panel's WMMAs + softmax. Online softmax; P re-striped via LDS.
// ---------------------------------------------------------------------------
__global__ __launch_bounds__(32) void attn_kernel(const __bf16* __restrict__ q_ws,
                                                  const __bf16* __restrict__ k_ws,
                                                  const __bf16* __restrict__ vt_ws,
                                                  __bf16* __restrict__ attn_ws) {
    __shared__ __attribute__((aligned(16))) __bf16 kbuf[2][32][EMB];  // 16 KB
    __shared__ __attribute__((aligned(16))) __bf16 ptile[16 * 32];    // 1 KB
    const int lane = threadIdx.x, half = lane >> 4, lm = lane & 15;
    const int it = blockIdx.x;     // 0..63
    const int h  = blockIdx.y;     // 0..7
    const int b  = blockIdx.z;     // 0..15
    const size_t bh = (size_t)(b * HH + h);
    const __bf16* Q  = q_ws  + (bh * NN + it * 16) * EMB;
    const __bf16* K  = k_ws  + bh * NN * EMB;
    const __bf16* Vt = vt_ws + bh * EMB * NN;

    // Q A-fragments (16x128 split into 4 k-steps of 32)
    v16bf aq[4];
    const __bf16* qrow = Q + (size_t)lm * EMB;
#pragma unroll
    for (int k0 = 0; k0 < 4; ++k0) aq[k0] = load_afrag(qrow + k0 * 32, half);

    float mrun[8], lrun[8];
    v8f o[8];
#pragma unroll
    for (int r = 0; r < 8; ++r) { mrun[r] = -1e30f; lrun[r] = 0.0f; o[r] = (v8f){}; }

    // Stage first K panel: 8192 B = 16 issues x (32 lanes x 16 B).
#pragma unroll
    for (int t = 0; t < 16; ++t) {
        int e = (t * 32 + lane) * 8;
        async_ld16(K + e, &kbuf[0][0][0] + e);
    }

    for (int js = 0; js < 32; ++js) {
        const int j0 = js * 32;
        const __bf16(*kc)[EMB] = kbuf[js & 1];
        if (js + 1 < 32) {
            const __bf16* src = K + (size_t)(j0 + 32) * EMB;
            __bf16* dst = &kbuf[(js + 1) & 1][0][0];
#pragma unroll
            for (int t = 0; t < 16; ++t) {
                int e = (t * 32 + lane) * 8;
                async_ld16(src + e, dst + e);
            }
            ASYNC_WAIT(16);   // current panel complete; next 16 still in flight
        } else {
            ASYNC_WAIT(0);
        }

        // two 16x16 score tiles from the LDS panel (ds_load_b128 B-fragments)
        v8f s0 = {}, s1 = {};
        {
            const __bf16* krow0 = &kc[lm][0];
            const __bf16* krow1 = &kc[16 + lm][0];
#pragma unroll
            for (int k0 = 0; k0 < 4; ++k0) {
                v16bf b0 = *(const v16bf*)(krow0 + k0 * 32 + 16 * half);
                v16bf b1 = *(const v16bf*)(krow1 + k0 * 32 + 16 * half);
                s0 = WMMA_BF16(aq[k0], b0, s0);
                s1 = WMMA_BF16(aq[k0], b1, s1);
            }
        }
        // online softmax per row (row = r + 8*half; cols live across 16 lanes)
        float alpha[8];
#pragma unroll
        for (int r = 0; r < 8; ++r) {
            float v0 = s0[r] * SCALE, v1 = s1[r] * SCALE;
            float mx = fmaxf(v0, v1);
            for (int off = 1; off < 16; off <<= 1) mx = fmaxf(mx, __shfl_xor(mx, off, 32));
            float mnew = fmaxf(mrun[r], mx);
            float p0 = __expf(v0 - mnew), p1 = __expf(v1 - mnew);
            float ps = p0 + p1;
            for (int off = 1; off < 16; off <<= 1) ps += __shfl_xor(ps, off, 32);
            alpha[r] = __expf(mrun[r] - mnew);
            lrun[r]  = lrun[r] * alpha[r] + ps;
            mrun[r]  = mnew;
            ptile[(r + 8 * half) * 32 + lm]      = (__bf16)p0;
            ptile[(r + 8 * half) * 32 + 16 + lm] = (__bf16)p1;
        }
        __syncthreads();
        // re-stripe P (C layout) into A layout: row m = lm, k = 32-wide
        v8bf plo = *(const v8bf*)(&ptile[lm * 32 + 16 * half]);
        v8bf phi = *(const v8bf*)(&ptile[lm * 32 + 16 * half + 8]);
        v16bf ap = cat8(plo, phi);
        __syncthreads();
        // O = O*alpha + P @ V   (8 d-tiles of 16; V fragments direct from global)
#pragma unroll
        for (int dt = 0; dt < 8; ++dt) {
            v8f oo = o[dt];
            for (int r = 0; r < 8; ++r) oo[r] *= alpha[r];
            const __bf16* vrow = Vt + (size_t)(dt * 16 + lm) * NN + j0;
            v16bf bv = *(const v16bf*)(vrow + 16 * half);
            o[dt] = WMMA_BF16(ap, bv, oo);
        }
    }
    // normalize + store bf16 into [b*N + i][h*128 + d]
#pragma unroll
    for (int dt = 0; dt < 8; ++dt) {
        for (int r = 0; r < 8; ++r) {
            int i = it * 16 + r + 8 * half;
            int d = dt * 16 + lm;
            float val = o[dt][r] / lrun[r];
            attn_ws[((size_t)(b * NN + i)) * INNER + h * EMB + d] = (__bf16)val;
        }
    }
}

// ---------------------------------------------------------------------------
// Kernel 5: output projection [16384 x 1024] @ [1024 x 128] + bias -> fp32.
// 16x64 C-tile per wave: 128 WMMAs per wave, A-fragment reused 4x per k-step.
// ---------------------------------------------------------------------------
__global__ __launch_bounds__(32) void proj_gemm(const __bf16* __restrict__ A,
                                                const __bf16* __restrict__ WpT,
                                                const float* __restrict__ bias,
                                                float* __restrict__ out) {
    const int lane = threadIdx.x, half = lane >> 4, lm = lane & 15;
    const int nt = blockIdx.x;     // 0..1 (64-wide N tiles)
    const int mt = blockIdx.y;     // 0..1023
    const __bf16* arow = A + (size_t)(mt * 16 + lm) * INNER;
    v8f acc[4] = {{}, {}, {}, {}};
    for (int k0 = 0; k0 < INNER; k0 += 32) {
        v16bf a = load_afrag(arow + k0, half);
#pragma unroll
        for (int c = 0; c < 4; ++c) {
            const __bf16* brow = WpT + (size_t)(nt * 64 + c * 16 + lm) * INNER;
            v16bf b = *(const v16bf*)(brow + k0 + 16 * half);
            acc[c] = WMMA_BF16(a, b, acc[c]);
        }
    }
#pragma unroll
    for (int c = 0; c < 4; ++c) {
        const float bn = bias[nt * 64 + c * 16 + lm];
        for (int r = 0; r < 8; ++r) {
            int gm = mt * 16 + r + 8 * half;
            out[(size_t)gm * EMB + nt * 64 + c * 16 + lm] = acc[c][r] + bn;
        }
    }
}

// ---------------------------------------------------------------------------
extern "C" void kernel_launch(void* const* d_in, const int* in_sizes, int n_in,
                              void* d_out, int out_size, void* d_ws, size_t ws_size,
                              hipStream_t stream) {
    (void)in_sizes; (void)n_in; (void)out_size; (void)ws_size;
    const float* x       = (const float*)d_in[0];
    const float* gamma   = (const float*)d_in[1];
    const float* beta    = (const float*)d_in[2];
    const float* w_qkv   = (const float*)d_in[3];
    const float* w_proj  = (const float*)d_in[4];
    const float* b_proj  = (const float*)d_in[5];
    float* out = (float*)d_out;

    char* ws = (char*)d_ws;
    size_t off = 0;
    __bf16* x_b    = (__bf16*)(ws + off); off += (size_t)BB * NN * EMB * 2;          // 4 MB
    __bf16* wqkv_t = (__bf16*)(ws + off); off += (size_t)QKVN * EMB * 2;             // 768 KB
    __bf16* wprj_t = (__bf16*)(ws + off); off += (size_t)EMB * INNER * 2;            // 256 KB
    __bf16* q_ws   = (__bf16*)(ws + off); off += (size_t)BB * HH * NN * EMB * 2;     // 32 MB
    __bf16* k_ws   = (__bf16*)(ws + off); off += (size_t)BB * HH * NN * EMB * 2;     // 32 MB
    __bf16* vt_ws  = (__bf16*)(ws + off); off += (size_t)BB * HH * NN * EMB * 2;     // 32 MB
    __bf16* a_ws   = (__bf16*)(ws + off);                                            // 32 MB

    ln_kernel<<<BB * NN, 32, 0, stream>>>(x, gamma, beta, x_b);
    conv_wqkv<<<(QKVN * EMB + 255) / 256, 256, 0, stream>>>(w_qkv, wqkv_t);
    conv_wproj<<<(EMB * INNER + 255) / 256, 256, 0, stream>>>(w_proj, wprj_t);
    qkv_gemm<<<dim3(QKVN / 64, BB * NN / 16), 32, 0, stream>>>(x_b, wqkv_t, q_ws, k_ws, vt_ws);
    attn_kernel<<<dim3(NN / 16, HH, BB), 32, 0, stream>>>(q_ws, k_ws, vt_ws, a_ws);
    proj_gemm<<<dim3(EMB / 64, BB * NN / 16), 32, 0, stream>>>(a_ws, wprj_t, b_proj, out);
}
